// JinaEmbeddingsV3SelfAttention_74809740362142
// MI455X (gfx1250) — matrix operand order repaired and checked by reading, not compiled, not run on hardware
//
#include <hip/hip_runtime.h>
#include <hip/hip_bf16.h>

// ---------------------------------------------------------------------------
// JinaEmbeddingsV3 self-attention (LoRA QKV + RoPE + softmax attention)
// MI455X / gfx1250: bf16 WMMA for both GEMM stages (f32 accumulate),
// Tensor Data Mover (tensor_load_to_lds) for shared-operand staging,
// f32 VALU for bias/LoRA/RoPE/softmax.
// ---------------------------------------------------------------------------

typedef __attribute__((ext_vector_type(16))) __bf16        v16bf;
typedef __attribute__((ext_vector_type(8)))  __bf16        v8bf;
typedef __attribute__((ext_vector_type(8)))  float         v8f;
typedef __attribute__((ext_vector_type(4)))  unsigned int  v4u;
typedef __attribute__((ext_vector_type(8)))  int           v8i;
typedef __attribute__((ext_vector_type(4)))  int           v4i;

#define NB   8
#define SEQ  1024
#define DIM  1024
#define NH   16
#define HD   64
#define OD   3072   // 3*DIM

static __device__ __forceinline__ unsigned short f2bf_rne(float f) {
  unsigned int u = __float_as_uint(f);
  unsigned int r = u + 0x7FFFu + ((u >> 16) & 1u);
  return (unsigned short)(r >> 16);
}

static __device__ __forceinline__ v16bf cat8(v8bf lo, v8bf hi) {
  return __builtin_shufflevector(lo, hi, 0,1,2,3,4,5,6,7,8,9,10,11,12,13,14,15);
}

static __device__ __forceinline__ v8f zero_v8f() {
  v8f z;
#pragma unroll
  for (int i = 0; i < 8; ++i) z[i] = 0.0f;
  return z;
}

// ---------------------------------------------------------------------------
// Tensor Data Mover: 2-D tile (data_size = 2 bytes) global -> LDS.
// D# group0/group1 per CDNA5 ISA 8.3/8.4; groups 2/3 zero (<=2D tensor).
// 1-D staging uses tile_d1 = 1.
// ---------------------------------------------------------------------------
static __device__ __forceinline__ void tdm_load_2d_bf16(
    unsigned lds_off, const void* gaddr,
    unsigned tile_d0, unsigned tile_d1,
    unsigned tensor_d0, unsigned tensor_d1, unsigned stride_d0) {
  unsigned long long ga = (unsigned long long)gaddr;
  v4u g0;
  g0[0] = 1u;                                            // count=1 (valid user D#)
  g0[1] = lds_off;                                       // lds_addr (bytes)
  g0[2] = (unsigned)(ga & 0xFFFFFFFFu);                  // global_addr[31:0]
  g0[3] = (unsigned)((ga >> 32) & 0x1FFFFFFu) | (2u << 30); // addr[56:32] | type=2
  v8i g1;
  g1[0] = (int)(1u << 16);                               // data_size=1 -> 2 bytes
  g1[1] = (int)((tensor_d0 & 0xFFFFu) << 16);            // tensor_dim0[15:0]
  g1[2] = (int)(((tensor_d0 >> 16) & 0xFFFFu) | ((tensor_d1 & 0xFFFFu) << 16));
  g1[3] = (int)(((tensor_d1 >> 16) & 0xFFFFu) | ((tile_d0 & 0xFFFFu) << 16));
  g1[4] = (int)(tile_d1 & 0xFFFFu);                      // tile_dim1 | tile_dim2=0
  g1[5] = (int)stride_d0;                                // tensor_dim0_stride[31:0]
  g1[6] = 0;
  g1[7] = 0;
  v4i z4; z4[0] = z4[1] = z4[2] = z4[3] = 0;
#if defined(__clang_major__) && (__clang_major__ >= 23)
  v8i z8;
#pragma unroll
  for (int i = 0; i < 8; ++i) z8[i] = 0;
  __builtin_amdgcn_tensor_load_to_lds(g0, g1, z4, z4, z8, 0);
#else
  __builtin_amdgcn_tensor_load_to_lds(g0, g1, z4, z4, 0);
#endif
}

static __device__ __forceinline__ unsigned lds_off_of(const void* p) {
  return (unsigned)(unsigned long long)p;   // generic LDS addr: low 32 bits = offset
}

// ---------------------------------------------------------------------------
// Kernel 1: f32 -> bf16 conversion (hidden states + Wqkv weight)
// ---------------------------------------------------------------------------
__global__ __launch_bounds__(256) void cvt_bf16_kernel(
    const float* __restrict__ a, const float* __restrict__ b,
    unsigned short* __restrict__ ao, unsigned short* __restrict__ bo,
    int na, int nbv) {
  long stride = (long)gridDim.x * blockDim.x;
  long total  = (long)na + nbv;
  for (long i = (long)blockIdx.x * blockDim.x + threadIdx.x; i < total; i += stride) {
    if (i < na) ao[i]      = f2bf_rne(a[i]);
    else        bo[i - na] = f2bf_rne(b[i - na]);
  }
}

// ---------------------------------------------------------------------------
// Kernel 2: LoRA down-projection lo[b,s,r] = sum_d hs[b,s,d]*A[am[b],r,d]
// ---------------------------------------------------------------------------
__global__ __launch_bounds__(256) void lora_lo_kernel(
    const float* __restrict__ hs, const float* __restrict__ loraA,
    const int* __restrict__ amask, float* __restrict__ lo) {
  int t = blockIdx.x * blockDim.x + threadIdx.x;   // 0..NB*SEQ-1
  int b = t >> 10;
  int am = amask[b];
  const float* x = hs + (long)t * DIM;
  const float* A = loraA + (long)am * 4 * DIM;
  float a0 = 0.f, a1 = 0.f, a2 = 0.f, a3 = 0.f;
  for (int d = 0; d < DIM; d += 4) {
    float4 xv = *(const float4*)(x + d);
    a0 += xv.x * A[d] + xv.y * A[d+1] + xv.z * A[d+2] + xv.w * A[d+3];
    a1 += xv.x * A[DIM+d] + xv.y * A[DIM+d+1] + xv.z * A[DIM+d+2] + xv.w * A[DIM+d+3];
    a2 += xv.x * A[2*DIM+d] + xv.y * A[2*DIM+d+1] + xv.z * A[2*DIM+d+2] + xv.w * A[2*DIM+d+3];
    a3 += xv.x * A[3*DIM+d] + xv.y * A[3*DIM+d+1] + xv.z * A[3*DIM+d+2] + xv.w * A[3*DIM+d+3];
  }
  *(float4*)(lo + (long)t * 4) = make_float4(a0, a1, a2, a3);
}

// ---------------------------------------------------------------------------
// Kernel 3: fused QKV GEMM + bias + LoRA delta + RoPE.
// Block = 1 M-tile (16 rows) x 8 N-tiles of 64 cols; the 16x1024 A-slab is
// shared by all 8 waves -> staged once via TDM (32KB, 1-D contiguous).
// ---------------------------------------------------------------------------
__global__ __launch_bounds__(256) void qkv_wmma_kernel(
    const unsigned short* __restrict__ hsb_u,
    const unsigned short* __restrict__ wb_u,
    const float* __restrict__ bias,
    const float* __restrict__ lo,
    const float* __restrict__ loraB,
    const int*   __restrict__ amask,
    const float* __restrict__ cosb,
    const float* __restrict__ sinb,
    unsigned short* __restrict__ qb,
    unsigned short* __restrict__ kb,
    unsigned short* __restrict__ vt) {
  __shared__ __align__(16) unsigned short ldsA[16 * DIM];   // 32 KB
  const __bf16* wb = (const __bf16*)wb_u;
  const int lane = threadIdx.x & 31;
  const int wave = threadIdx.x >> 5;
  const int ln = lane & 15, g = lane >> 4;

  const int mt  = blockIdx.x / 6;           // 0..511
  const int ntg = blockIdx.x - mt * 6;      // 0..5
  const int row0 = mt * 16;                 // base row in [B*S]
  const int bIdx = row0 >> 10;
  const int obase = ntg * 512 + wave * 64;  // 64-aligned output col base

  if (wave == 0) {
    tdm_load_2d_bf16(lds_off_of(ldsA), hsb_u + (long)row0 * DIM,
                     /*tile*/ 16 * DIM, 1,
                     /*tensor*/ 1u << 20, 1u << 20, /*stride*/ 16 * DIM);
    __builtin_amdgcn_s_wait_tensorcnt(0);
  }
  __syncthreads();

  const __bf16* Al = (const __bf16*)ldsA + (long)ln * DIM;
  const __bf16* Bp[4];
#pragma unroll
  for (int sub = 0; sub < 4; ++sub)
    Bp[sub] = wb + (long)(obase + sub * 16 + ln) * DIM + g * 16;

  v8f c[4];
#pragma unroll
  for (int i = 0; i < 4; ++i) c[i] = zero_v8f();

  for (int k = 0; k < DIM; k += 32) {
    __builtin_prefetch(Bp[0] + k + 512, 0, 0);     // global_prefetch_b8
    v16bf a = cat8(*(const v8bf*)(Al + k + g * 16),
                   *(const v8bf*)(Al + k + g * 16 + 8));
#pragma unroll
    for (int sub = 0; sub < 4; ++sub) {
      v16bf bm = *(const v16bf*)(Bp[sub] + k);
      c[sub] = __builtin_amdgcn_wmma_f32_16x16x32_bf16(
          false, a, false, bm, (short)0, c[sub], false, false);
    }
  }

  // ---- epilogue: + bias + LoRA delta ----
  const int am = amask[bIdx];
  const float* Bt = loraB + (long)am * OD * 4;

  float l4x[8], l4y[8], l4z[8], l4w[8];
#pragma unroll
  for (int r = 0; r < 8; ++r) {
    int grow = row0 + r + 8 * g;
    float4 lr = *(const float4*)(lo + (long)grow * 4);
    l4x[r] = lr.x; l4y[r] = lr.y; l4z[r] = lr.z; l4w[r] = lr.w;
  }

  float v[4][8];
#pragma unroll
  for (int sub = 0; sub < 4; ++sub) {
    int o = obase + sub * 16 + ln;
    float4 bt = *(const float4*)(Bt + (long)o * 4);
    float bo = bias[o];
#pragma unroll
    for (int r = 0; r < 8; ++r)
      v[sub][r] = c[sub][r] + bo +
                  0.25f * (l4x[r] * bt.x + l4y[r] * bt.y +
                           l4z[r] * bt.z + l4w[r] * bt.w);
  }

  const int idx = obase >> 10;            // 0=q 1=k 2=v (wave-uniform)
  const int h   = (obase & 1023) >> 6;

  if (idx < 2) {
    // RoPE: pair (d, d+32) = (sub, sub+2) lives in the same lane.
    unsigned short* dst = (idx == 0) ? qb : kb;
    long base = (long)(bIdx * NH + h) * SEQ * HD;
#pragma unroll
    for (int sub = 0; sub < 2; ++sub) {
      int dlo = sub * 16 + ln, dhi = dlo + 32;
#pragma unroll
      for (int r = 0; r < 8; ++r) {
        int srow = (row0 + r + 8 * g) & (SEQ - 1);
        long cs = ((long)bIdx * SEQ + srow) * HD;
        float cl = cosb[cs + dlo], sl = sinb[cs + dlo];
        float ch = cosb[cs + dhi], sh = sinb[cs + dhi];
        float xlo = v[sub][r], xhi = v[sub + 2][r];
        dst[base + (long)srow * HD + dlo] = f2bf_rne(xlo * cl - xhi * sl);
        dst[base + (long)srow * HD + dhi] = f2bf_rne(xhi * ch + xlo * sh);
      }
    }
  } else {
    // v stored transposed: vT[b,h,d,s] so PV B-operand is contiguous.
    long base = (long)(bIdx * NH + h) * HD * SEQ;
#pragma unroll
    for (int sub = 0; sub < 4; ++sub) {
      int d = sub * 16 + ln;
#pragma unroll
      for (int r = 0; r < 8; ++r) {
        int srow = (row0 + r + 8 * g) & (SEQ - 1);
        vt[base + (long)d * SEQ + srow] = f2bf_rne(v[sub][r]);
      }
    }
  }
}

// ---------------------------------------------------------------------------
// Kernel 4: flash attention. Block = one (b,h,128 q rows); all 8 waves share
// each 32-key K/V chunk -> double-buffered TDM staging in LDS; online softmax;
// P re-laid out C->A through a private per-wave LDS slab.
// ---------------------------------------------------------------------------
__global__ __launch_bounds__(256) void flash_attn_kernel(
    const unsigned short* __restrict__ qb_u,
    const unsigned short* __restrict__ kb_u,
    const unsigned short* __restrict__ vt_u,
    const float* __restrict__ mask,
    float* __restrict__ out) {
  __shared__ __align__(16) unsigned short bufK[2][32 * HD];   // 2 x 4 KB, [kp][d]
  __shared__ __align__(16) unsigned short bufV[2][HD * 32];   // 2 x 4 KB, [d][kp]
  __shared__ __align__(16) unsigned short pbuf[8][16 * 32];   // 8 KB P slabs
  const __bf16* qb = (const __bf16*)qb_u;

  const int lane = threadIdx.x & 31;
  const int wave = threadIdx.x >> 5;
  const int ln = lane & 15, g = lane >> 4;
  const int bh   = blockIdx.x >> 3;        // (b*16+h)
  const int qblk = blockIdx.x & 7;
  const int b = bh >> 4, h = bh & 15;
  const int qbase = qblk * 128 + wave * 16;
  const float scale = 0.125f;              // 64^-0.5
  const long hb = (long)bh * SEQ * HD;

  const __bf16* qp = qb + hb + (long)(qbase + ln) * HD;
  v16bf qa0 = cat8(*(const v8bf*)(qp + g * 16),      *(const v8bf*)(qp + 8 + g * 16));
  v16bf qa1 = cat8(*(const v8bf*)(qp + 32 + g * 16), *(const v8bf*)(qp + 40 + g * 16));

  float m[8], l[8];
  v8f o[4];
#pragma unroll
  for (int r = 0; r < 8; ++r) { m[r] = -1e30f; l[r] = 0.0f; }
#pragma unroll
  for (int dt = 0; dt < 4; ++dt) o[dt] = zero_v8f();

  const float* mrow = mask + (long)b * SEQ;
  unsigned short* pw = pbuf[wave];

  // prologue: stage chunk 0 (K: contiguous 32x64; V: 2-D 64x32, stride SEQ)
  if (wave == 0) {
    tdm_load_2d_bf16(lds_off_of(bufK[0]), kb_u + hb, 32 * HD, 1,
                     1u << 20, 1u << 20, 32 * HD);
    tdm_load_2d_bf16(lds_off_of(bufV[0]), vt_u + hb, 32, HD,
                     1u << 20, 1u << 20, SEQ);
  }

  for (int it = 0; it < SEQ / 32; ++it) {
    const int kp = it * 32;
    const int cb = it & 1;
    if (wave == 0) {
      if (it + 1 < SEQ / 32) {
        tdm_load_2d_bf16(lds_off_of(bufK[cb ^ 1]),
                         kb_u + hb + (long)(kp + 32) * HD, 32 * HD, 1,
                         1u << 20, 1u << 20, 32 * HD);
        tdm_load_2d_bf16(lds_off_of(bufV[cb ^ 1]),
                         vt_u + hb + (kp + 32), 32, HD,
                         1u << 20, 1u << 20, SEQ);
        __builtin_amdgcn_s_wait_tensorcnt(2);   // older pair done, newer in flight
      } else {
        __builtin_amdgcn_s_wait_tensorcnt(0);
      }
    }
    __syncthreads();    // chunk `it` resident; prev buffer free for TDM overwrite

    const __bf16* Kl = (const __bf16*)bufK[cb];
    const __bf16* Vl = (const __bf16*)bufV[cb];

    // --- scores: two 16-col tiles, K=64 as 2x32 ---
    v8f s0 = zero_v8f(), s1 = zero_v8f();
    s0 = __builtin_amdgcn_wmma_f32_16x16x32_bf16(false, qa0, false, *(const v16bf*)(Kl + ln * HD + g * 16),             (short)0, s0, false, false);
    s0 = __builtin_amdgcn_wmma_f32_16x16x32_bf16(false, qa1, false, *(const v16bf*)(Kl + ln * HD + 32 + g * 16),        (short)0, s0, false, false);
    s1 = __builtin_amdgcn_wmma_f32_16x16x32_bf16(false, qa0, false, *(const v16bf*)(Kl + (16 + ln) * HD + g * 16),      (short)0, s1, false, false);
    s1 = __builtin_amdgcn_wmma_f32_16x16x32_bf16(false, qa1, false, *(const v16bf*)(Kl + (16 + ln) * HD + 32 + g * 16), (short)0, s1, false, false);

    float msk0 = mrow[kp + ln], msk1 = mrow[kp + 16 + ln];
#pragma unroll
    for (int r = 0; r < 8; ++r) {
      float p0 = s0[r] * scale + msk0;
      float p1 = s1[r] * scale + msk1;
      float t = fmaxf(p0, p1);
      t = fmaxf(t, __shfl_xor(t, 1));
      t = fmaxf(t, __shfl_xor(t, 2));
      t = fmaxf(t, __shfl_xor(t, 4));
      t = fmaxf(t, __shfl_xor(t, 8));      // row max within 16-lane half
      float mn = fmaxf(m[r], t);
      float alpha = __expf(m[r] - mn);
      p0 = __expf(p0 - mn);
      p1 = __expf(p1 - mn);
      float rs = p0 + p1;
      rs += __shfl_xor(rs, 1);
      rs += __shfl_xor(rs, 2);
      rs += __shfl_xor(rs, 4);
      rs += __shfl_xor(rs, 8);
      l[r] = l[r] * alpha + rs;
      m[r] = mn;
#pragma unroll
      for (int dt = 0; dt < 4; ++dt) o[dt][r] *= alpha;
      int Mr = r + 8 * g;
      pw[Mr * 32 + ln]      = f2bf_rne(p0);
      pw[Mr * 32 + 16 + ln] = f2bf_rne(p1);
    }
    // intra-wave C->A relayout through private slab: DS ops are in-order per
    // wave; the fence only stops compiler reordering of the LDS accesses.
    asm volatile("" ::: "memory");
    v16bf pa = cat8(*(const v8bf*)(pw + ln * 32 + g * 16),
                    *(const v8bf*)(pw + ln * 32 + g * 16 + 8));
#pragma unroll
    for (int dt = 0; dt < 4; ++dt) {
      o[dt] = __builtin_amdgcn_wmma_f32_16x16x32_bf16(
          false, pa, false, *(const v16bf*)(Vl + (dt * 16 + ln) * 32 + g * 16),
          (short)0, o[dt], false, false);
    }
    __syncthreads();    // all reads of buf[cb] done before TDM reuses it
  }

#pragma unroll
  for (int dt = 0; dt < 4; ++dt)
#pragma unroll
    for (int r = 0; r < 8; ++r) {
      int sq = qbase + r + 8 * g;
      out[((long)b * SEQ + sq) * (NH * HD) + h * HD + dt * 16 + ln] =
          o[dt][r] / l[r];
    }
}

// ---------------------------------------------------------------------------
extern "C" void kernel_launch(void* const* d_in, const int* in_sizes, int n_in,
                              void* d_out, int out_size, void* d_ws, size_t ws_size,
                              hipStream_t stream) {
  const float* hs    = (const float*)d_in[0];
  const float* mask  = (const float*)d_in[1];
  const float* cosb  = (const float*)d_in[2];
  const float* sinb  = (const float*)d_in[3];
  const int*   amask = (const int*)d_in[4];
  const float* W     = (const float*)d_in[5];
  const float* bias  = (const float*)d_in[6];
  const float* loraA = (const float*)d_in[7];
  const float* loraB = (const float*)d_in[8];
  float* out = (float*)d_out;

  char* ws = (char*)d_ws;
  const long n_hs = (long)NB * SEQ * DIM;   // 8388608
  const long n_w  = (long)OD * DIM;         // 3145728
  unsigned short* hsb = (unsigned short*)ws;  ws += n_hs * 2;
  unsigned short* wb  = (unsigned short*)ws;  ws += n_w * 2;
  float*          lo  = (float*)ws;           ws += (long)NB * SEQ * 4 * 4;
  unsigned short* qb  = (unsigned short*)ws;  ws += (long)NB * NH * SEQ * HD * 2;
  unsigned short* kb  = (unsigned short*)ws;  ws += (long)NB * NH * SEQ * HD * 2;
  unsigned short* vt  = (unsigned short*)ws;

  cvt_bf16_kernel<<<4096, 256, 0, stream>>>(hs, W, hsb, wb, (int)n_hs, (int)n_w);
  lora_lo_kernel<<<(NB * SEQ) / 256, 256, 0, stream>>>(hs, loraA, amask, lo);
  qkv_wmma_kernel<<<512 * 6, 256, 0, stream>>>(
      hsb, wb, bias, lo, loraB, amask, cosb, sinb, qb, kb, vt);
  flash_attn_kernel<<<NB * NH * (SEQ / 128), 256, 0, stream>>>(
      qb, kb, vt, mask, out);
}
